// Attention_43894565765586
// MI455X (gfx1250) — matrix operand order repaired and checked by reading, not compiled, not run on hardware
//
#include <hip/hip_runtime.h>
#include <hip/hip_bf16.h>

typedef unsigned short u16;
typedef __attribute__((ext_vector_type(16))) __bf16        v16bf;
typedef __attribute__((ext_vector_type(8)))  float         v8f;
typedef __attribute__((ext_vector_type(4)))  unsigned int  u32x4;
typedef __attribute__((ext_vector_type(8)))  int           i32x8;
typedef __attribute__((ext_vector_type(4)))  int           i32x4;
typedef __attribute__((ext_vector_type(2)))  float         f32x2;

#if defined(__has_builtin)
#if __has_builtin(__builtin_amdgcn_tensor_load_to_lds)
#define ATHENA_TDM 1
#endif
#endif
#ifndef ATHENA_TDM
#define ATHENA_TDM 0
#endif

union BfTile {
  v16bf v;
  u32x4 q[2];
  unsigned int w[8];
  u16 h[16];
};

// ---------- helpers ----------

__device__ __forceinline__ u16 f2bf(float f) {
  unsigned int u = __float_as_uint(f);
  u += 0x7fffu + ((u >> 16) & 1u);      // round-to-nearest-even
  return (u16)(u >> 16);
}

// A-operand (16xK tile, 16-bit): lane holds row; contiguous 16 at kb+hi*16
__device__ __forceinline__ v16bf load_a16(const u16* rowbase, int kbase, int hi) {
  BfTile t;
  t.q[0] = *(const u32x4*)(rowbase + kbase + hi * 16);
  t.q[1] = *(const u32x4*)(rowbase + kbase + hi * 16 + 8);
  return t.v;
}
// B-operand (Kx16 tile, 16-bit, source row-major over K): contiguous 16 at kb+hi*16
__device__ __forceinline__ v16bf load_b16(const u16* rowbase, int kbase, int hi) {
  BfTile t;
  const u16* p = rowbase + kbase + hi * 16;
  t.q[0] = *(const u32x4*)(p);
  t.q[1] = *(const u32x4*)(p + 8);
  return t.v;
}

__device__ __forceinline__ v8f wmma_bf16(v16bf a, v16bf b, v8f c) {
  return __builtin_amdgcn_wmma_f32_16x16x32_bf16(false, a, false, b, (short)0, c,
                                                 false, false);
}

#if ATHENA_TDM
// 2D tile (tile_d1 rows x tile_d0 elems of 2 bytes) from global -> LDS via TDM.
// D# fields per CDNA5 ISA 8.3/8.4: group0 = {count, lds_addr, global_addr, type=2},
// group1 = {data_size=1(2B), tensor_dim0/1, tile_dim0/1, tensor_dim0_stride}.
__device__ __forceinline__ void tdm_load_2d(unsigned lds_off, const void* gaddr,
                                            unsigned tile_d0, unsigned tile_d1,
                                            unsigned tensor_d0, unsigned tensor_d1,
                                            unsigned stride_d0_elems) {
  unsigned long long ga = (unsigned long long)(uintptr_t)gaddr;
  u32x4 g0;
  g0[0] = 1u;                                           // count=1 (valid), user mode
  g0[1] = lds_off;                                      // lds_addr (bytes)
  g0[2] = (unsigned)(ga & 0xffffffffu);                 // global_addr[31:0]
  g0[3] = (unsigned)((ga >> 32) & 0x01ffffffu) | (2u << 30);  // addr[56:32] | type=2
  i32x8 g1;
  g1[0] = (int)(1u << 16);                              // data_size = 1 -> 2 bytes
  g1[1] = (int)((tensor_d0 & 0xffffu) << 16);           // tensor_dim0 low16 (bits 63:48)
  g1[2] = (int)(((tensor_d0 >> 16) & 0xffffu) | ((tensor_d1 & 0xffffu) << 16));
  g1[3] = (int)(((tensor_d1 >> 16) & 0xffffu) | ((tile_d0 & 0xffffu) << 16));
  g1[4] = (int)(tile_d1 & 0xffffu);                     // tile_dim1, tile_dim2=0
  g1[5] = (int)stride_d0_elems;                         // tensor_dim0_stride[31:0]
  g1[6] = 0;
  g1[7] = 0;
  i32x4 z4 = {};
#if __has_include(<hip/amd_detail/amd_gfx1250_TDM.h>)
  i32x8 z8 = {};
  __builtin_amdgcn_tensor_load_to_lds(g0, g1, z4, z4, z8, 0);
#else
  __builtin_amdgcn_tensor_load_to_lds(g0, g1, z4, z4, 0);
#endif
}
#endif

// ---------- conversion / transpose kernels ----------

__global__ void f2bf_kernel(const float* __restrict__ in, u16* __restrict__ out, int n) {
  int i = blockIdx.x * blockDim.x + threadIdx.x;
  if (i < n) out[i] = f2bf(in[i]);
}

// W[K][N] fp32 -> Wt[N][K] bf16
__global__ void wtrans(const float* __restrict__ W, u16* __restrict__ Wt, int K, int N) {
  __shared__ float tile[16][17];
  int n0 = blockIdx.x * 16, k0 = blockIdx.y * 16;
  tile[threadIdx.y][threadIdx.x] = W[(size_t)(k0 + threadIdx.y) * N + n0 + threadIdx.x];
  __syncthreads();
  Wt[(size_t)(n0 + threadIdx.y) * K + k0 + threadIdx.x] = f2bf(tile[threadIdx.x][threadIdx.y]);
}

// vlin fp32 [B*T][4096] -> Vt bf16 [B][64][64][1024]  (head-major, DH rows, T cols)
__global__ void v_transpose(const float* __restrict__ vlin, u16* __restrict__ Vt) {
  __shared__ float tile[16][17];
  int bh = blockIdx.z;
  int b = bh >> 6, h = bh & 63;
  int t0 = blockIdx.x * 16, d0 = blockIdx.y * 16;
  tile[threadIdx.y][threadIdx.x] =
      vlin[(size_t)(b * 1024 + t0 + threadIdx.y) * 4096 + h * 64 + d0 + threadIdx.x];
  __syncthreads();
  Vt[((size_t)bh * 64 + d0 + threadIdx.y) * 1024 + t0 + threadIdx.x] =
      f2bf(tile[threadIdx.x][threadIdx.y]);
}

// ---------- bf16 WMMA GEMM:  C[M][N] = A[M][K] @ Bt[N][K]^T + bias ----------
// Block = 4 waves, tile 128m x 64n; wave owns 32m x 64n (8 accumulators).
// Shared 64x32 B panel staged in LDS (TDM double-buffered when available).

__global__ __launch_bounds__(128) void gemm_bf16(
    const u16* __restrict__ A, const u16* __restrict__ Bt,
    const float* __restrict__ bias, float* __restrict__ C,
    int M, int N, int K)
{
  __shared__ u16 sB[2][64 * 32];
  const int tid   = threadIdx.x;
  const int lane  = tid & 31;
  const int laneL = lane & 15;
  const int hi    = lane >> 4;
  const int wave  = tid >> 5;
  const int mbase = blockIdx.y * 128 + wave * 32;
  const int nbase = blockIdx.x * 64;

  const u16* arow0 = A + (size_t)(mbase + laneL) * K;
  const u16* arow1 = arow0 + (size_t)16 * K;

  v8f acc[2][4] = {};

#if ATHENA_TDM
  if (wave == 0) {
    tdm_load_2d((unsigned)(uintptr_t)&sB[0][0], Bt + (size_t)nbase * K,
                32u, 64u, (unsigned)K, (unsigned)N, (unsigned)K);
    __builtin_amdgcn_s_wait_tensorcnt(0);
  }
#else
  {
    int row = tid >> 1, half = tid & 1;
    const u32x4* src = (const u32x4*)(Bt + (size_t)(nbase + row) * K + half * 16);
    u32x4* dst = (u32x4*)&sB[0][row * 32 + half * 16];
    dst[0] = src[0]; dst[1] = src[1];
  }
#endif
  __syncthreads();

  for (int kc = 0; kc < K; kc += 32) {
    const int cur = (kc >> 5) & 1;
    const int nxt = cur ^ 1;
    const bool more = (kc + 32) < K;
#if ATHENA_TDM
    if (more && wave == 0)
      tdm_load_2d((unsigned)(uintptr_t)&sB[nxt][0],
                  Bt + (size_t)nbase * K + kc + 32,
                  32u, 64u, (unsigned)K, (unsigned)N, (unsigned)K);
#else
    const int row = tid >> 1, half = tid & 1;
    u32x4 pf0 = {}, pf1 = {};
    if (more) {
      const u32x4* src =
          (const u32x4*)(Bt + (size_t)(nbase + row) * K + kc + 32 + half * 16);
      pf0 = src[0]; pf1 = src[1];
    }
#endif
    __builtin_prefetch(arow0 + kc + 256, 0, 0);       // global_prefetch_b8
    __builtin_prefetch(arow1 + kc + 256, 0, 0);
    v16bf a0 = load_a16(arow0, kc, hi);
    v16bf a1 = load_a16(arow1, kc, hi);
    #pragma unroll
    for (int nt = 0; nt < 4; ++nt) {
      v16bf bb = load_b16(&sB[cur][(nt * 16 + laneL) * 32], 0, hi);
      acc[0][nt] = wmma_bf16(a0, bb, acc[0][nt]);
      acc[1][nt] = wmma_bf16(a1, bb, acc[1][nt]);
    }
#if ATHENA_TDM
    if (more && wave == 0) __builtin_amdgcn_s_wait_tensorcnt(0);
#else
    if (more) {
      u32x4* dst = (u32x4*)&sB[nxt][row * 32 + half * 16];
      dst[0] = pf0; dst[1] = pf1;
    }
#endif
    __syncthreads();
  }

  #pragma unroll
  for (int sub = 0; sub < 2; ++sub)
    #pragma unroll
    for (int nt = 0; nt < 4; ++nt) {
      int col = nbase + nt * 16 + laneL;
      float bs = bias[col];
      #pragma unroll
      for (int r = 0; r < 8; ++r) {
        int rowm = mbase + sub * 16 + hi * 8 + r;
        C[(size_t)rowm * N + col] = acc[sub][nt][r] + bs;
      }
    }
}

// ---------- Q post: rms_norm over DH then rope; fp32 -> bf16 [B][64][T][64] ----------

__global__ __launch_bounds__(256) void q_post(const float* __restrict__ qlin,
                                              u16* __restrict__ Qbf) {
  const int lane = threadIdx.x & 31;
  const int wave = threadIdx.x >> 5;
  const int r  = blockIdx.x * 8 + wave;          // [0, B*64*T)
  const int t  = r & 1023;
  const int bh = r >> 10;                        // b*64 + h
  const int b  = bh >> 6, h = bh & 63;
  f32x2 x = *(const f32x2*)(qlin + (size_t)(b * 1024 + t) * 4096 + h * 64 + lane * 2);
  float ss = x.x * x.x + x.y * x.y;
  #pragma unroll
  for (int msk = 1; msk < 32; msk <<= 1) ss += __shfl_xor(ss, msk, 32);
  const float inv = rsqrtf(ss * (1.f / 64.f) + 1.1920929e-7f);
  const float x1 = x.x * inv, x2 = x.y * inv;
  float sn, cs;
  __sincosf((float)t * __powf(10000.f, -(float)lane * (1.f / 32.f)), &sn, &cs);
  u16* o = Qbf + ((size_t)bh * 1024 + t) * 64;
  o[lane]      = f2bf(x1 * cs - x2 * sn);        // first half  (i)
  o[lane + 32] = f2bf(x1 * sn + x2 * cs);        // second half (32+i)
}

// ---------- K post: rope + denom=sqrt(sum k^2); bf16 [B][16][T][64] ----------

__global__ __launch_bounds__(256) void k_post(const float* __restrict__ klin,
                                              u16* __restrict__ Kbf,
                                              float* __restrict__ denomK) {
  const int lane = threadIdx.x & 31;
  const int wave = threadIdx.x >> 5;
  const int r  = blockIdx.x * 8 + wave;          // [0, B*16*T)
  const int t  = r & 1023;
  const int bs = r >> 10;                        // b*16 + sh
  f32x2 x = *(const f32x2*)(klin + (size_t)((bs >> 4) * 1024 + t) * 1024 +
                            (bs & 15) * 64 + lane * 2);
  float ss = x.x * x.x + x.y * x.y;
  #pragma unroll
  for (int msk = 1; msk < 32; msk <<= 1) ss += __shfl_xor(ss, msk, 32);
  if (lane == 0) denomK[(size_t)bs * 1024 + t] = sqrtf(ss);  // rope preserves norm
  float sn, cs;
  __sincosf((float)t * __powf(10000.f, -(float)lane * (1.f / 32.f)), &sn, &cs);
  u16* o = Kbf + ((size_t)bs * 1024 + t) * 64;
  o[lane]      = f2bf(x.x * cs - x.y * sn);
  o[lane + 32] = f2bf(x.x * sn + x.y * cs);
}

// ---------- flash attention with sink, transposed-score WMMA ----------
// grid (T/128, 64 heads, B), block 256 (8 waves); wave owns 16 query rows.
// K tile (32x64) and V^T tile (64x32) staged in LDS, shared by all 8 waves.

__global__ __launch_bounds__(256) void attn(
    const u16* __restrict__ Qbf, const u16* __restrict__ Kbf,
    const u16* __restrict__ Vt,  const float* __restrict__ denomK,
    const float* __restrict__ sink_scalars, const float* __restrict__ v_nulls,
    float* __restrict__ O)
{
  __shared__ u16 sK[2][32 * 64];
  __shared__ u16 sV[2][64 * 32];
  const int tid   = threadIdx.x;
  const int lane  = tid & 31;
  const int laneL = lane & 15;
  const int hi    = lane >> 4;
  const int wave  = tid >> 5;
  const int h = blockIdx.y, b = blockIdx.z;
  const int br = h >> 4, sh = h & 15;
  const int WH[4] = {16, 128, 128, 512};
  const int WG[4] = {0, 16, 0, 144};
  const int wh = WH[br], wg = WG[br];
  const int q  = blockIdx.x * 128 + wave * 16 + laneL;

  const u16* qrow = Qbf + ((size_t)(b * 64 + h) * 1024 + q) * 64;
  const v16bf bq0 = load_b16(qrow, 0, hi);       // Q^T as B-operand, dh 0..31
  const v16bf bq1 = load_b16(qrow, 32, hi);      // dh 32..63

  const u16*  kbasep = Kbf + (size_t)(b * 16 + sh) * 1024 * 64;
  const float* dptr  = denomK + (size_t)(b * 16 + sh) * 1024;
  const u16*  vbase  = Vt + (size_t)(b * 64 + h) * 64 * 1024;

  const float sinkv = tanhf(sink_scalars[h]);
  float m = sinkv;                               // sink folded into softmax init
  float l = 1.f;
  v8f acc[4] = {};

#if ATHENA_TDM
  if (wave == 0) {
    tdm_load_2d((unsigned)(uintptr_t)&sK[0][0], kbasep, 64u, 32u, 64u, 1024u, 64u);
    tdm_load_2d((unsigned)(uintptr_t)&sV[0][0], vbase,  32u, 64u, 1024u, 64u, 1024u);
    __builtin_amdgcn_s_wait_tensorcnt(0);
  }
#else
  {
    int kr = tid >> 3, ks = tid & 7;             // 32 rows x 8 segs of 8 halfwords
    *(u32x4*)&sK[0][kr * 64 + ks * 8] = *(const u32x4*)(kbasep + (size_t)kr * 64 + ks * 8);
    int vr = tid >> 2, vs = tid & 3;             // 64 rows x 4 segs
    *(u32x4*)&sV[0][vr * 32 + vs * 8] = *(const u32x4*)(vbase + (size_t)vr * 1024 + vs * 8);
  }
#endif
  __syncthreads();

  for (int kc = 0; kc < 1024; kc += 32) {
    const int cur = (kc >> 5) & 1;
    const int nxt = cur ^ 1;
    const bool more = kc + 32 < 1024;
#if ATHENA_TDM
    if (more && wave == 0) {
      tdm_load_2d((unsigned)(uintptr_t)&sK[nxt][0], kbasep + (size_t)(kc + 32) * 64,
                  64u, 32u, 64u, 1024u, 64u);
      tdm_load_2d((unsigned)(uintptr_t)&sV[nxt][0], vbase + kc + 32,
                  32u, 64u, 1024u, 64u, 1024u);
    }
#else
    const int kr = tid >> 3, ks = tid & 7;
    const int vr = tid >> 2, vs = tid & 3;
    u32x4 pk = {}, pv = {};
    if (more) {
      pk = *(const u32x4*)(kbasep + (size_t)(kc + 32 + kr) * 64 + ks * 8);
      pv = *(const u32x4*)(vbase + (size_t)vr * 1024 + kc + 32 + vs * 8);
    }
#endif

    // S^T = K . Q^T : lane = query column, VGPR = key row
    v8f s0 = {}, s1 = {};
    {
      const u16* kr0 = &sK[cur][laneL * 64];
      const u16* kr1 = &sK[cur][(16 + laneL) * 64];
      s0 = wmma_bf16(load_a16(kr0, 0, hi),  bq0, s0);
      s0 = wmma_bf16(load_a16(kr0, 32, hi), bq1, s0);
      s1 = wmma_bf16(load_a16(kr1, 0, hi),  bq0, s1);
      s1 = wmma_bf16(load_a16(kr1, 32, hi), bq1, s1);
    }

    const int k0i = kc + hi * 8;
    const int k1i = kc + 16 + hi * 8;
    float p0[8], p1[8];
    float lm = -3.4e38f;
    #pragma unroll
    for (int r = 0; r < 8; ++r) {
      int ka = k0i + r, kb = k1i + r;
      int da = q - ka, db = q - kb;
      float la = (da >= wg && da < wh) ? s0[r] / dptr[ka] : 0.f;  // mask -> logit 0
      float lb = (db >= wg && db < wh) ? s1[r] / dptr[kb] : 0.f;
      p0[r] = la; p1[r] = lb;
      lm = fmaxf(lm, fmaxf(la, lb));
    }
    lm = fmaxf(lm, __shfl_xor(lm, 16, 32));      // merge key halves (same q)
    const float mnew  = fmaxf(m, lm);
    const float scale = __expf(m - mnew);
    float lsum = 0.f;
    #pragma unroll
    for (int r = 0; r < 8; ++r) {
      p0[r] = __expf(p0[r] - mnew);
      p1[r] = __expf(p1[r] - mnew);
      lsum += p0[r] + p1[r];
    }
    l = l * scale + lsum + __shfl_xor(lsum, 16, 32);
    m = mnew;
    #pragma unroll
    for (int a4 = 0; a4 < 4; ++a4)
      #pragma unroll
      for (int r = 0; r < 8; ++r) acc[a4][r] *= scale;

    // Repack P^T (C-layout) into B-operand striping: only shfl_xor(16) needed.
    BfTile bp;
    #pragma unroll
    for (int j = 0; j < 4; ++j) {
      unsigned int t0 = (unsigned int)f2bf(p0[2 * j]) |
                        ((unsigned int)f2bf(p0[2 * j + 1]) << 16);
      unsigned int t1 = (unsigned int)f2bf(p1[2 * j]) |
                        ((unsigned int)f2bf(p1[2 * j + 1]) << 16);
      unsigned int e0 = (unsigned int)__shfl_xor((int)t0, 16, 32);
      unsigned int e1 = (unsigned int)__shfl_xor((int)t1, 16, 32);
      bp.w[j]     = hi ? e1 : t0;   // K 0..7  (hi=0) | K 16..23 (hi=1)
      bp.w[4 + j] = hi ? t1 : e0;   // K 8..15 (hi=0 from partner) | K 24..31
    }

    // out^T += V^T . P^T
    #pragma unroll
    for (int dt = 0; dt < 4; ++dt)
      acc[dt] = wmma_bf16(load_a16(&sV[cur][(dt * 16 + laneL) * 32], 0, hi),
                          bp.v, acc[dt]);

#if ATHENA_TDM
    if (more && wave == 0) __builtin_amdgcn_s_wait_tensorcnt(0);
#else
    if (more) {
      *(u32x4*)&sK[nxt][kr * 64 + ks * 8] = pk;
      *(u32x4*)&sV[nxt][vr * 32 + vs * 8] = pv;
    }
#endif
    __syncthreads();
  }

  const float invl = 1.f / l;
  const float sw = __expf(sinkv - m) * invl;     // sink probability
  float* orow = O + ((size_t)(b * 64 + h) * 1024 + q) * 64;
  #pragma unroll
  for (int dt = 0; dt < 4; ++dt) {
    int d0 = dt * 16 + hi * 8;
    #pragma unroll
    for (int r = 0; r < 8; ++r)
      orow[d0 + r] = acc[dt][r] * invl + sw * v_nulls[h * 64 + d0 + r];
  }
}

// ---------- branch mean:  ctx_bf16[b][t][sh*64+d] = 0.25 * sum_br O[b][br*16+sh][t][d]

__global__ void reduce_branches(const float* __restrict__ O, u16* __restrict__ ctx) {
  int idx = blockIdx.x * blockDim.x + threadIdx.x;   // < 2*1024*1024
  int c = idx & 1023;
  int t = (idx >> 10) & 1023;
  int b = idx >> 20;
  int sh = c >> 6, d = c & 63;
  float s = 0.f;
  #pragma unroll
  for (int br = 0; br < 4; ++br)
    s += O[((size_t)(b * 64 + br * 16 + sh) * 1024 + t) * 64 + d];
  ctx[idx] = f2bf(0.25f * s);
}

// ---------- launch ----------

extern "C" void kernel_launch(void* const* d_in, const int* in_sizes, int n_in,
                              void* d_out, int out_size, void* d_ws, size_t ws_size,
                              hipStream_t stream) {
  const float* X    = (const float*)d_in[0];
  const float* WQ   = (const float*)d_in[1];
  const float* bQ   = (const float*)d_in[2];
  const float* WK   = (const float*)d_in[3];
  const float* bK   = (const float*)d_in[4];
  const float* WV   = (const float*)d_in[5];
  const float* bV   = (const float*)d_in[6];
  const float* WO   = (const float*)d_in[7];
  const float* bO   = (const float*)d_in[8];
  const float* sink = (const float*)d_in[9];
  const float* vnul = (const float*)d_in[10];
  float* Y = (float*)d_out;

  char* base = (char*)d_ws;
  size_t off = 0;
  auto alloc = [&](size_t bytes) -> void* {
    void* p = base + off;
    off = (off + bytes + 255) & ~(size_t)255;
    return p;
  };

  const size_t M = 2048;                          // B*T
  u16*   Xbf   = (u16*)  alloc(M * 1024 * 2);
  u16*   WQt   = (u16*)  alloc((size_t)4096 * 1024 * 2);
  u16*   WKt   = (u16*)  alloc((size_t)1024 * 1024 * 2);
  u16*   WVt   = (u16*)  alloc((size_t)4096 * 1024 * 2);
  u16*   WOt   = (u16*)  alloc((size_t)1024 * 1024 * 2);
  float* qlin  = (float*)alloc(M * 4096 * 4);
  float* klin  = (float*)alloc(M * 1024 * 4);
  float* vlin  = (float*)alloc(M * 4096 * 4);
  u16*   Qbf   = (u16*)  alloc((size_t)2 * 64 * 1024 * 64 * 2);
  u16*   Kbf   = (u16*)  alloc((size_t)2 * 16 * 1024 * 64 * 2);
  float* denomK= (float*)alloc((size_t)2 * 16 * 1024 * 4);
  u16*   Vtb   = (u16*)  alloc((size_t)2 * 64 * 64 * 1024 * 2);
  float* Obuf  = (float*)alloc((size_t)2 * 64 * 1024 * 64 * 4);
  u16*   ctx   = (u16*)  alloc(M * 1024 * 2);
  (void)ws_size; (void)in_sizes; (void)n_in; (void)out_size;

  dim3 tb(16, 16);
  f2bf_kernel<<<8192, 256, 0, stream>>>(X, Xbf, 2 * 1024 * 1024);
  wtrans<<<dim3(256, 64), tb, 0, stream>>>(WQ, WQt, 1024, 4096);
  wtrans<<<dim3(64, 64),  tb, 0, stream>>>(WK, WKt, 1024, 1024);
  wtrans<<<dim3(256, 64), tb, 0, stream>>>(WV, WVt, 1024, 4096);
  wtrans<<<dim3(64, 64),  tb, 0, stream>>>(WO, WOt, 1024, 1024);

  gemm_bf16<<<dim3(64, 16), 128, 0, stream>>>(Xbf, WQt, bQ, qlin, 2048, 4096, 1024);
  gemm_bf16<<<dim3(16, 16), 128, 0, stream>>>(Xbf, WKt, bK, klin, 2048, 1024, 1024);
  gemm_bf16<<<dim3(64, 16), 128, 0, stream>>>(Xbf, WVt, bV, vlin, 2048, 4096, 1024);

  q_post<<<2 * 64 * 1024 / 8, 256, 0, stream>>>(qlin, Qbf);
  k_post<<<2 * 16 * 1024 / 8, 256, 0, stream>>>(klin, Kbf, denomK);
  v_transpose<<<dim3(64, 4, 128), tb, 0, stream>>>(vlin, Vtb);

  attn<<<dim3(8, 64, 2), 256, 0, stream>>>(Qbf, Kbf, Vtb, denomK, sink, vnul, Obuf);

  reduce_branches<<<8192, 256, 0, stream>>>(Obuf, ctx);
  gemm_bf16<<<dim3(16, 16), 128, 0, stream>>>(ctx, WOt, bO, Y, 2048, 1024, 1024);
}
